// Attention_88905823027347
// MI455X (gfx1250) — matrix operand order repaired and checked by reading, not compiled, not run on hardware
//
#include <hip/hip_runtime.h>

typedef __attribute__((ext_vector_type(16))) _Float16 v16h;
typedef __attribute__((ext_vector_type(8)))  _Float16 v8h;
typedef __attribute__((ext_vector_type(8)))  float    v8f;
typedef __attribute__((ext_vector_type(4)))  unsigned u32x4;
typedef __attribute__((ext_vector_type(8)))  int      i32x8;
typedef __attribute__((ext_vector_type(4)))  int      i32x4;

#define BB    2
#define NN    4096
#define CC    512
#define HEADS 8
#define DH    64
#define NKV   1024
#define RR    8

// ---- Tensor Data Mover availability (arity differs per toolchain) ----------
#if __has_include(<hip/amd_detail/amd_gfx1250_TDM.h>)
  #define ATH_HAVE_TDM 1
  #define ATH_TDM_6ARG 1
#elif __has_builtin(__builtin_amdgcn_tensor_load_to_lds)
  #define ATH_HAVE_TDM 1
#endif

static __device__ __forceinline__ v8f wmma16(v16h a, v16h b, v8f c) {
  return __builtin_amdgcn_wmma_f32_16x16x32_f16(false, a, false, b, (short)0, c,
                                                false, false);
}

// Build a v16h from two 16B-aligned 8-half loads.
static __device__ __forceinline__ v16h ld2x8(const _Float16* p, const _Float16* q) {
  union { v16h v; v8h h[2]; } u;
  u.h[0] = *(const v8h*)p;
  u.h[1] = *(const v8h*)q;
  return u.v;
}

// A-operand K index for lane-half g, element i (16-bit A 16x32 layout).
static __device__ __forceinline__ int a_kidx(int i, int g) {
  return (i < 8) ? (i + 8 * g) : (i + 8 + 8 * g);
}

#ifdef ATH_HAVE_TDM
// TDM: DMA a [rows x 64] f16 tile (row stride 512 elements) into LDS,
// padding 4 DWORDs every 32 DWORDs -> LDS row pitch = 72 halves (144B).
static __device__ __forceinline__ void tdm_load_tile_2d(unsigned lds_off,
                                                        const void* gptr,
                                                        int rows) {
  unsigned long long ga = (unsigned long long)(size_t)gptr;
  u32x4 g0;
  g0[0] = 1u;                                   // count=1 (user descriptor)
  g0[1] = lds_off;                              // lds_addr
  g0[2] = (unsigned)ga;                         // global_addr[31:0]
  g0[3] = (unsigned)((ga >> 32) & 0x1FFFFFFu) | (2u << 30);  // addr[56:32]|type=2
  i32x8 g1;
  g1[0] = (1 << 16) | (1 << 20) | (4 << 22) | (3 << 25); // 2B elems, pad 4dw/32dw
  g1[1] = (int)(512u << 16);                    // tensor_dim0 = 512
  g1[2] = (int)(1024u << 16);                   // tensor_dim1 = 1024
  g1[3] = (int)(64u << 16);                     // tile_dim0 = 64
  g1[4] = rows;                                 // tile_dim1
  g1[5] = 512;                                  // tensor_dim0_stride = 512
  g1[6] = 0;
  g1[7] = 0;
  i32x4 z4 = {0, 0, 0, 0};
#ifdef ATH_TDM_6ARG
  i32x8 z8 = {0, 0, 0, 0, 0, 0, 0, 0};
  __builtin_amdgcn_tensor_load_to_lds(g0, g1, z4, z4, z8, 0);
#else
  __builtin_amdgcn_tensor_load_to_lds(g0, g1, z4, z4, 0);
#endif
}
#endif

// ---------------------------------------------------------------- conversions
__global__ void cvt_f32_f16(const float* __restrict__ in, _Float16* __restrict__ out, int n) {
  int idx = blockIdx.x * blockDim.x + threadIdx.x;
  if (idx < n) out[idx] = (_Float16)in[idx];
}

// Fold LoRA into effective weights, convert to f16, store TRANSPOSED [n][k].
__global__ void make_eff_w(const float* __restrict__ Wq, const float* __restrict__ Wkv,
                           const float* __restrict__ Wproj,
                           const float* __restrict__ Aq, const float* __restrict__ Bq,
                           const float* __restrict__ Av, const float* __restrict__ Bv,
                           _Float16* __restrict__ wqT, _Float16* __restrict__ wkT,
                           _Float16* __restrict__ wvT, _Float16* __restrict__ wpT) {
  int idx = blockIdx.x * blockDim.x + threadIdx.x;   // 512*512
  int i = idx >> 9, j = idx & 511;                   // i = k (input), j = n (output)
  float lq = 0.f, lv = 0.f;
#pragma unroll
  for (int r = 0; r < RR; ++r) {
    lq += Aq[i * RR + r] * Bq[r * CC + j];
    lv += Av[i * RR + r] * Bv[r * CC + j];
  }
  wqT[j * CC + i] = (_Float16)(Wq[idx] + lq);
  wkT[j * CC + i] = (_Float16)(Wkv[i * (2 * CC) + j] + lv);
  wvT[j * CC + i] = (_Float16)(Wkv[i * (2 * CC) + CC + j] + lv);
  wpT[j * CC + i] = (_Float16)Wproj[idx];
}

// Wsr (2048 x 512 in HWIO-flat) -> transposed f16 [co][k]
__global__ void cvt_wsr_t(const float* __restrict__ in, _Float16* __restrict__ outT) {
  int idx = blockIdx.x * blockDim.x + threadIdx.x;   // 2048*512
  int kk = idx >> 9, co = idx & 511;
  outT[(size_t)co * 2048 + kk] = (_Float16)in[idx];
}

// im2col for the 2x2 stride-2 conv: patch[row=(b,oi,oj)][col=h*1024+w*512+ci]
__global__ void build_patch(const _Float16* __restrict__ xh, _Float16* __restrict__ patch) {
  int idx = blockIdx.x * blockDim.x + threadIdx.x;   // 2048 * 2048
  int row = idx >> 11, col = idx & 2047;
  int b  = row >> 10;
  int oi = (row >> 5) & 31;
  int oj = row & 31;
  int hh = col >> 10;
  int ww = (col >> 9) & 1;
  int ci = col & 511;
  int src = ((b * NN) + (2 * oi + hh) * 64 + (2 * oj + ww)) * CC + ci;
  patch[idx] = xh[src];
}

// ------------------------------------------------------------------ layernorm
__global__ void layernorm_rows(const float* __restrict__ in, const float* __restrict__ gamma,
                               const float* __restrict__ beta, _Float16* __restrict__ out) {
  int row  = blockIdx.x * blockDim.y + threadIdx.y;   // blockDim = (32, 8)
  int lane = threadIdx.x;
  const float* p = in + (size_t)row * CC;
  float vals[16], s = 0.f, ss = 0.f;
#pragma unroll
  for (int i = 0; i < 16; ++i) {
    float v = p[lane + 32 * i];
    vals[i] = v; s += v; ss += v * v;
  }
#pragma unroll
  for (int m = 1; m < 32; m <<= 1) {
    s  += __shfl_xor(s,  m, 32);
    ss += __shfl_xor(ss, m, 32);
  }
  float mu  = s * (1.0f / CC);
  float var = ss * (1.0f / CC) - mu * mu;
  float inv = rsqrtf(var + 1e-5f);
#pragma unroll
  for (int i = 0; i < 16; ++i) {
    int c = lane + 32 * i;
    out[(size_t)row * CC + c] = (_Float16)((vals[i] - mu) * inv * gamma[c] + beta[c]);
  }
}

// ------------------------------------------- LDS-free 16x64 WMMA GEMM (B^T)
// out = A(MxK) @ B(KxN) + bias, with B supplied transposed Bt[N][K].
// Both WMMA operands are contiguous 16-half runs -> pure b128 global loads.
__global__ __launch_bounds__(32)
void gemm_16x64_tn(const _Float16* __restrict__ A, const _Float16* __restrict__ Bt,
                   const float* __restrict__ bias, float* __restrict__ outF,
                   _Float16* __restrict__ outH, int N, int K) {
  int tilesN = N >> 6;
  int m0 = (blockIdx.x / tilesN) * 16;
  int n0 = (blockIdx.x % tilesN) * 64;
  int lane = threadIdx.x, g = lane >> 4, c0 = lane & 15;

  const _Float16* arow = A  + (size_t)(m0 + c0) * K;
  const _Float16* br0  = Bt + (size_t)(n0 + c0) * K;
  const _Float16* br1  = Bt + (size_t)(n0 + 16 + c0) * K;
  const _Float16* br2  = Bt + (size_t)(n0 + 32 + c0) * K;
  const _Float16* br3  = Bt + (size_t)(n0 + 48 + c0) * K;

  v8f acc0 = {0,0,0,0,0,0,0,0}, acc1 = acc0, acc2 = acc0, acc3 = acc0;

#pragma unroll 4
  for (int k0 = 0; k0 < K; k0 += 32) {
    int ka = k0 + 16 * g;       // A: contiguous 16 per lane-half
    int kb = k0 + 16 * g;       // B: contiguous 16 per lane-half
    v16h a  = ld2x8(arow + ka, arow + ka + 8);
    v16h b0 = ld2x8(br0 + kb, br0 + kb + 8);
    v16h b1 = ld2x8(br1 + kb, br1 + kb + 8);
    v16h b2 = ld2x8(br2 + kb, br2 + kb + 8);
    v16h b3 = ld2x8(br3 + kb, br3 + kb + 8);
    acc0 = wmma16(a, b0, acc0);
    acc1 = wmma16(a, b1, acc1);
    acc2 = wmma16(a, b2, acc2);
    acc3 = wmma16(a, b3, acc3);
  }

  float bb0 = bias[n0 + c0];
  float bb1 = bias[n0 + 16 + c0];
  float bb2 = bias[n0 + 32 + c0];
  float bb3 = bias[n0 + 48 + c0];
#pragma unroll
  for (int r = 0; r < 8; ++r) {
    int m = m0 + r + 8 * g;                        // D layout: M = r + 8*(lane>>4)
    size_t ro = (size_t)m * N + n0;
    float v0 = acc0[r] + bb0, v1 = acc1[r] + bb1;
    float v2 = acc2[r] + bb2, v3 = acc3[r] + bb3;
    if (outF) {
      outF[ro + c0] = v0; outF[ro + 16 + c0] = v1;
      outF[ro + 32 + c0] = v2; outF[ro + 48 + c0] = v3;
    }
    if (outH) {
      outH[ro + c0] = (_Float16)v0; outH[ro + 16 + c0] = (_Float16)v1;
      outH[ro + 32 + c0] = (_Float16)v2; outH[ro + 48 + c0] = (_Float16)v3;
    }
  }
}

// --------------------------------------------------------- flash attention
// One wave per (b, h, 16-query tile); streams 1024 keys in blocks of 32.
// K blocks staged to LDS via the Tensor Data Mover when available.
__global__ __launch_bounds__(32)
void flash_attn(const _Float16* __restrict__ qf, const _Float16* __restrict__ kf,
                const _Float16* __restrict__ vf, _Float16* __restrict__ of) {
  int tile = blockIdx.x;            // B * HEADS * (NN/16) = 4096
  int qt = tile & 255;
  int h  = (tile >> 8) & 7;
  int b  = tile >> 11;
  int lane = threadIdx.x, g = lane >> 4, c0 = lane & 15;

  __shared__ _Float16 sK[32 * 72];   // row-major [key][dh], pitch 72 (TDM pad)
  __shared__ _Float16 sVt[64 * 48];  // transposed [dh][key]
  __shared__ _Float16 sP[16 * 40];

  const _Float16* qbase = qf + ((size_t)(b * NN + qt * 16)) * CC + h * DH;
  const _Float16* kbase = kf + ((size_t)(b * NKV)) * CC + h * DH;
  const _Float16* vbase = vf + ((size_t)(b * NKV)) * CC + h * DH;

  // Q tile straight from global into A-operand layout, pre-scaled by dh^-0.5.
  const _Float16* qrow = qbase + (size_t)c0 * CC;
  v16h aq0 = ld2x8(qrow + 16 * g,      qrow + 8 + 16 * g);
  v16h aq1 = ld2x8(qrow + 32 + 16 * g, qrow + 40 + 16 * g);
  const float scale = 0.125f;
#pragma unroll
  for (int i = 0; i < 16; ++i) {
    aq0[i] = (_Float16)((float)aq0[i] * scale);
    aq1[i] = (_Float16)((float)aq1[i] * scale);
  }

  float mrow[8], lrow[8];
  v8f O0 = {0,0,0,0,0,0,0,0}, O1 = O0, O2 = O0, O3 = O0;
#pragma unroll
  for (int r = 0; r < 8; ++r) { mrow[r] = -1e30f; lrow[r] = 0.f; }

  for (int kb = 0; kb < NKV; kb += 32) {
    __syncthreads();   // previous iteration's LDS reads fully done
#ifdef ATH_HAVE_TDM
    tdm_load_tile_2d((unsigned)(size_t)&sK[0], kbase + (size_t)kb * CC, 32);
#else
#pragma unroll
    for (int t = 0; t < 8; ++t) {
      int j = lane + 32 * t, r = j >> 3, seg = j & 7;
      *(uint4*)(&sK[r * 72 + seg * 8]) =
          *(const uint4*)(kbase + (size_t)(kb + r) * CC + seg * 8);
    }
#endif
    // V block 32x64, stored transposed [dh][key]
#pragma unroll
    for (int t = 0; t < 8; ++t) {
      int j = lane + 32 * t, r = j >> 3, seg = (j & 7) * 8;
      uint4 pk = *(const uint4*)(vbase + (size_t)(kb + r) * CC + seg);
      const _Float16* hp = (const _Float16*)&pk;
#pragma unroll
      for (int s = 0; s < 8; ++s) sVt[(seg + s) * 48 + r] = hp[s];
    }
    if (kb + 32 < NKV) {   // prefetch next blocks (uniform condition)
      __builtin_prefetch(kbase + (size_t)(kb + 32 + lane) * CC, 0, 1);
      __builtin_prefetch(vbase + (size_t)(kb + 32 + lane) * CC, 0, 1);
    }
#ifdef ATH_HAVE_TDM
    __builtin_amdgcn_s_wait_tensorcnt(0);
#endif
    __syncthreads();

    // S = (scale*Q) @ K^T : two 16x16 tiles, dh accumulated in two 32-chunks.
    v8f S0 = {0,0,0,0,0,0,0,0}, S1 = S0;
    S0 = wmma16(aq0, ld2x8(&sK[(c0)*72 + 16*g],        &sK[(c0)*72 + 16*g + 8]),      S0);
    S0 = wmma16(aq1, ld2x8(&sK[(c0)*72 + 32 + 16*g],   &sK[(c0)*72 + 32 + 16*g + 8]), S0);
    S1 = wmma16(aq0, ld2x8(&sK[(16+c0)*72 + 16*g],     &sK[(16+c0)*72 + 16*g + 8]),   S1);
    S1 = wmma16(aq1, ld2x8(&sK[(16+c0)*72 + 32 + 16*g],&sK[(16+c0)*72 + 32+16*g + 8]),S1);

    // Online softmax (rows of a tile live in one 16-lane half)
#pragma unroll
    for (int r = 0; r < 8; ++r) {
      float x0 = S0[r], x1 = S1[r];
      float mx = fmaxf(x0, x1);
#pragma unroll
      for (int m = 1; m < 16; m <<= 1) mx = fmaxf(mx, __shfl_xor(mx, m, 32));
      float mnew  = fmaxf(mrow[r], mx);
      float alpha = __expf(mrow[r] - mnew);
      float p0 = __expf(x0 - mnew), p1 = __expf(x1 - mnew);
      float rs = p0 + p1;
#pragma unroll
      for (int m = 1; m < 16; m <<= 1) rs += __shfl_xor(rs, m, 32);
      lrow[r] = lrow[r] * alpha + rs;
      mrow[r] = mnew;
      O0[r] *= alpha; O1[r] *= alpha; O2[r] *= alpha; O3[r] *= alpha;
      int mr = r + 8 * g;
      sP[mr * 40 + c0]      = (_Float16)p0;
      sP[mr * 40 + 16 + c0] = (_Float16)p1;
    }
    __syncthreads();

    // O += P(16x32) @ V(32x64)
    v16h ap;
#pragma unroll
    for (int i = 0; i < 16; ++i) ap[i] = sP[c0 * 40 + 16 * g + i];
    O0 = wmma16(ap, ld2x8(&sVt[(c0)*48 + 16*g],    &sVt[(c0)*48 + 16*g + 8]),    O0);
    O1 = wmma16(ap, ld2x8(&sVt[(16+c0)*48 + 16*g], &sVt[(16+c0)*48 + 16*g + 8]), O1);
    O2 = wmma16(ap, ld2x8(&sVt[(32+c0)*48 + 16*g], &sVt[(32+c0)*48 + 16*g + 8]), O2);
    O3 = wmma16(ap, ld2x8(&sVt[(48+c0)*48 + 16*g], &sVt[(48+c0)*48 + 16*g + 8]), O3);
  }

#pragma unroll
  for (int r = 0; r < 8; ++r) {
    float inv = 1.0f / lrow[r];
    int m = r + 8 * g;
    size_t ro = ((size_t)(b * NN + qt * 16 + m)) * CC + h * DH;
    of[ro + c0]      = (_Float16)(O0[r] * inv);
    of[ro + 16 + c0] = (_Float16)(O1[r] * inv);
    of[ro + 32 + c0] = (_Float16)(O2[r] * inv);
    of[ro + 48 + c0] = (_Float16)(O3[r] * inv);
  }
}

// ------------------------------------------------------------------- launch
extern "C" void kernel_launch(void* const* d_in, const int* in_sizes, int n_in,
                              void* d_out, int out_size, void* d_ws, size_t ws_size,
                              hipStream_t stream) {
  (void)in_sizes; (void)n_in; (void)out_size; (void)ws_size;
  const float* x     = (const float*)d_in[0];
  const float* Wq    = (const float*)d_in[1];
  const float* bq    = (const float*)d_in[2];
  const float* Wkv   = (const float*)d_in[3];
  const float* bkv   = (const float*)d_in[4];
  const float* Wproj = (const float*)d_in[5];
  const float* bproj = (const float*)d_in[6];
  const float* Aq    = (const float*)d_in[7];
  const float* Bq    = (const float*)d_in[8];
  const float* Av    = (const float*)d_in[9];
  const float* Bv    = (const float*)d_in[10];
  const float* Wsr   = (const float*)d_in[11];
  const float* bsr   = (const float*)d_in[12];
  const float* gamma = (const float*)d_in[13];
  const float* beta  = (const float*)d_in[14];
  float* out = (float*)d_out;
  char*  ws  = (char*)d_ws;

  // workspace layout (bytes, all 256-aligned)
  const size_t OFF_XH    = 0;            // 8192*512 f16
  const size_t OFF_PATCH = 8388608;      // 2048*2048 f16
  const size_t OFF_Q     = 16777216;     // 8192*512 f16
  const size_t OFF_AO    = 25165824;     // 8192*512 f16
  const size_t OFF_WQ    = 33554432;     // 512*512 f16 (transposed)
  const size_t OFF_WK    = 34078720;
  const size_t OFF_WV    = 34603008;
  const size_t OFF_WP    = 35127296;
  const size_t OFF_WSR   = 35651584;     // 512*2048 f16 (transposed)
  const size_t OFF_XSF   = 37748736;     // 2048*512 f32
  const size_t OFF_XSH   = 41943040;     // 2048*512 f16
  const size_t OFF_K     = 44040192;
  const size_t OFF_V     = 46137344;     // end = 48234496 (~46 MB)

  _Float16* xh    = (_Float16*)(ws + OFF_XH);
  _Float16* patch = (_Float16*)(ws + OFF_PATCH);
  _Float16* qh    = (_Float16*)(ws + OFF_Q);
  _Float16* aoh   = (_Float16*)(ws + OFF_AO);
  _Float16* wqT   = (_Float16*)(ws + OFF_WQ);
  _Float16* wkT   = (_Float16*)(ws + OFF_WK);
  _Float16* wvT   = (_Float16*)(ws + OFF_WV);
  _Float16* wpT   = (_Float16*)(ws + OFF_WP);
  _Float16* wsrT  = (_Float16*)(ws + OFF_WSR);
  float*    xsf   = (float*)   (ws + OFF_XSF);
  _Float16* xsh   = (_Float16*)(ws + OFF_XSH);
  _Float16* kh    = (_Float16*)(ws + OFF_K);
  _Float16* vh    = (_Float16*)(ws + OFF_V);

  // 1. conversions / weight folding (weights stored transposed)
  cvt_f32_f16<<<16384, 256, 0, stream>>>(x, xh, BB * NN * CC);
  cvt_wsr_t<<<4096, 256, 0, stream>>>(Wsr, wsrT);
  make_eff_w<<<1024, 256, 0, stream>>>(Wq, Wkv, Wproj, Aq, Bq, Av, Bv,
                                       wqT, wkT, wvT, wpT);
  build_patch<<<16384, 256, 0, stream>>>(xh, patch);

  // 2. q projection (LoRA folded):  8192x512 @ 512x512
  gemm_16x64_tn<<<(BB * NN / 16) * (CC / 64), 32, 0, stream>>>(
      xh, wqT, bq, nullptr, qh, CC, CC);

  // 3. conv-as-GEMM:  2048x2048 @ 2048x512
  gemm_16x64_tn<<<(BB * NKV / 16) * (CC / 64), 32, 0, stream>>>(
      patch, wsrT, bsr, xsf, nullptr, CC, 4 * CC);

  // 4. LayerNorm
  layernorm_rows<<<BB * NKV / 8, dim3(32, 8), 0, stream>>>(xsf, gamma, beta, xsh);

  // 5. k / v projections (LoRA folded)
  gemm_16x64_tn<<<(BB * NKV / 16) * (CC / 64), 32, 0, stream>>>(
      xsh, wkT, bkv, nullptr, kh, CC, CC);
  gemm_16x64_tn<<<(BB * NKV / 16) * (CC / 64), 32, 0, stream>>>(
      xsh, wvT, bkv + CC, nullptr, vh, CC, CC);

  // 6. flash attention (TDM-staged K blocks when toolchain supports it)
  flash_attn<<<BB * HEADS * (NN / 16), 32, 0, stream>>>(qh, kh, vh, aoh);

  // 7. output projection -> f32 d_out
  gemm_16x64_tn<<<(BB * NN / 16) * (CC / 64), 32, 0, stream>>>(
      aoh, wpT, bproj, out, nullptr, CC, CC);
}
